// BackboneVaDER_63711544869459
// MI455X (gfx1250) — compile-verified, matrix operand order checked by reading, not run on hardware
//
#include <hip/hip_runtime.h>
#include <hip/hip_bf16.h>
#include <math.h>

// ---------------- problem dims ----------------
#define BB   256
#define TT   128
#define DD   128
#define HH   512
#define DMU  64
#define KK   8
#define FOURH 2048   // 4*H
#define THREEH 1536  // 3*H
#define BT   (BB*TT) // 32768

// ---------------- WMMA types ----------------
typedef __attribute__((ext_vector_type(16))) __bf16 v16bf;
typedef __attribute__((ext_vector_type(8)))  float  v8f;
typedef __attribute__((ext_vector_type(4)))  unsigned uint4e;

union FragBF { v16bf v; uint4e q[2]; };

__device__ inline unsigned short f2bf(float f) {
    unsigned u = __builtin_bit_cast(unsigned, f);
    unsigned r = u + 0x7FFFu + ((u >> 16) & 1u);   // round-to-nearest-even
    return (unsigned short)(r >> 16);
}

__device__ inline float softplus_f(float x) {
    return (x > 0.f) ? (x + log1pf(expf(-x))) : log1pf(expf(x));
}

__device__ inline float sigmoid_f(float x) { return 1.f / (1.f + expf(-x)); }

// Load a 16x32 bf16 tile (A-operand layout; identical layout serves the
// B-operand of X@W^T when W is row-major [N x K]).
// lane 0..15 : row = lane,     K = [k0..k0+7]  and [k0+16..k0+23]
// lane 16..31: row = lane-16,  K = [k0+8..k0+15] and [k0+24..k0+31]
__device__ inline v16bf ld_frag(const unsigned short* base, int stride, int lane, int k0) {
    int r  = lane & 15;
    int kb = k0 + ((lane >> 4) << 3);
    const unsigned short* p = base + (size_t)r * stride + kb;
    FragBF f;
    f.q[0] = *reinterpret_cast<const uint4e*>(p);
    f.q[1] = *reinterpret_cast<const uint4e*>(p + 16);
    return f.v;
}

__device__ inline v8f wmma_bf16(v16bf a, v16bf b, v8f c) {
    return __builtin_amdgcn_wmma_f32_16x16x32_bf16(false, a, false, b,
                                                   (short)0, c, false, false);
}

#define ZERO8 {0.f,0.f,0.f,0.f,0.f,0.f,0.f,0.f}

// ---------------- prep: fp32 -> bf16 ----------------
__global__ void k_f32_to_bf16(const float* __restrict__ src,
                              unsigned short* __restrict__ dst, int n) {
    int i = blockIdx.x * blockDim.x + threadIdx.x;
    if (i < n) dst[i] = f2bf(src[i]);
}

// ---------------- kernel 1: imputation ----------------
// Ximp = X*mask + (X @ impW^T) * (1-mask), store bf16
// M=BT, N=D=128, K=D=128. One wave = 16 rows x all 128 cols (A reused 8x).
__global__ void k_imp_gemm(const unsigned short* __restrict__ Xbf,
                           const unsigned short* __restrict__ Wbf,
                           const float* __restrict__ X,
                           const float* __restrict__ mask,
                           unsigned short* __restrict__ XimpBf) {
    const int wv = threadIdx.x >> 5, lane = threadIdx.x & 31;
    const int row0 = (blockIdx.x * 8 + wv) * 16;
    v8f acc[8] = {ZERO8,ZERO8,ZERO8,ZERO8,ZERO8,ZERO8,ZERO8,ZERO8};
    for (int k0 = 0; k0 < DD; k0 += 32) {
        v16bf a = ld_frag(Xbf + (size_t)row0 * DD, DD, lane, k0);
#pragma unroll
        for (int i = 0; i < 8; ++i) {
            v16bf b = ld_frag(Wbf + (size_t)(i * 16) * DD, DD, lane, k0);
            acc[i] = wmma_bf16(a, b, acc[i]);
        }
    }
    const int mbase = (lane >> 4) << 3, nn = lane & 15;
#pragma unroll
    for (int i = 0; i < 8; ++i)
        for (int j = 0; j < 8; ++j) {
            size_t gi = (size_t)(row0 + mbase + j) * DD + i * 16 + nn;
            float x = X[gi], mk = mask[gi];
            XimpBf[gi] = f2bf(x * mk + acc[i][j] * (1.f - mk));
        }
}

// ---------------- kernel 2: wx_all = Ximp @ encWih^T + bih ----------------
// M=BT, N=4H=2048, K=D=128 -> fp32 workspace [BT, 4H]
// One wave = 16 rows x 128-col group (A reused 8x).
__global__ void k_wx_gemm(const unsigned short* __restrict__ A,
                          const unsigned short* __restrict__ W,
                          const float* __restrict__ bih,
                          float* __restrict__ wx) {
    const int wv = threadIdx.x >> 5, lane = threadIdx.x & 31;
    const int job = blockIdx.x * 8 + wv;           // 32768 jobs
    const int tm = job >> 4, ng = job & 15;        // 2048 m-tiles x 16 n-groups
    const int row0 = tm * 16, nbase = ng * 128;
    v8f acc[8] = {ZERO8,ZERO8,ZERO8,ZERO8,ZERO8,ZERO8,ZERO8,ZERO8};
    for (int k0 = 0; k0 < DD; k0 += 32) {
        v16bf a = ld_frag(A + (size_t)row0 * DD, DD, lane, k0);
#pragma unroll
        for (int i = 0; i < 8; ++i) {
            v16bf b = ld_frag(W + (size_t)(nbase + i * 16) * DD, DD, lane, k0);
            acc[i] = wmma_bf16(a, b, acc[i]);
        }
    }
    const int mbase = (lane >> 4) << 3, nn = lane & 15;
#pragma unroll
    for (int i = 0; i < 8; ++i) {
        const int col = nbase + i * 16 + nn;
        const float bv = bih[col];
        for (int j = 0; j < 8; ++j)
            wx[(size_t)(row0 + mbase + j) * FOURH + col] = acc[i][j] + bv;
    }
}

// ---------------- kernel 3/5: peephole LSTM recurrence ----------------
// One workgroup owns 16 batch rows for all T steps. State in LDS.
// Tiles are interleaved across waves (n0 = (tt*8+wv)*16) so the peephole
// predicate is compile-time per tile index and all waves do 12/16 peep tiles.
__global__ void k_lstm_rec(const unsigned short* __restrict__ Whh,  // [4H, H] bf16
                           const unsigned short* __restrict__ Wch,  // [3H, H] bf16
                           const float* __restrict__ bhh,           // [4H]
                           const float* __restrict__ bch,           // [3H]
                           const float* __restrict__ wx_all,        // [B,T,4H] or null
                           const float* __restrict__ bih_dec,       // [4H] or null
                           const float* __restrict__ h_init,        // [B,H] or null
                           float* __restrict__ cT_out,              // [B,H] or null
                           unsigned short* __restrict__ hs_out,     // [B,T,H] bf16 or null
                           int T_steps) {
    extern __shared__ char smem[];
    float*          hS = (float*)smem;                    // 16*512 f32
    float*          cS = hS + 16 * HH;                    // 16*512 f32
    unsigned short* hB = (unsigned short*)(cS + 16 * HH); // 16*512 bf16
    unsigned short* cB = hB + 16 * HH;                    // 16*512 bf16
    float*          g4 = (float*)(cB + 16 * HH);          // 16*2048 f32

    const int tid = threadIdx.x;
    const int wv = tid >> 5, lane = tid & 31;
    const int b0 = blockIdx.x * 16;
    const int mbase = (lane >> 4) << 3;
    const int nn = lane & 15;

    for (int idx = tid; idx < 16 * HH; idx += blockDim.x) {
        int r = idx >> 9, hc = idx & (HH - 1);
        float hv = h_init ? h_init[(b0 + r) * HH + hc] : 0.f;
        hS[idx] = hv; cS[idx] = 0.f;
        hB[idx] = f2bf(hv); cB[idx] = 0;
    }
    __syncthreads();

    for (int t = 0; t < T_steps; ++t) {
        // --- matmul phase: g4 = wx + h@Whh^T + bhh (+ c@Wch^T + bch on i/f/o) ---
#pragma unroll
        for (int pass = 0; pass < 2; ++pass) {
            v8f acc[8];
            // init from wx (+ biases)
#pragma unroll
            for (int i = 0; i < 8; ++i) {
                const int tt = pass * 8 + i;
                const int n0 = (tt * 8 + wv) * 16;
                const bool peep = (pass == 0) || (i >= 4);   // tt<8 || tt>=12
                const int c0 = (n0 < 2 * HH) ? n0 : (n0 - HH);
                const int col = n0 + nn;
                const float bias = bhh[col] + (peep ? bch[c0 + nn] : 0.f);
                if (wx_all) {
                    const float* wxp = wx_all +
                        ((size_t)(b0 + mbase) * (size_t)T_steps + t) * FOURH + col;
                    for (int j = 0; j < 8; ++j)
                        acc[i][j] = wxp[(size_t)j * T_steps * FOURH] + bias;
                } else {
                    const float wxv = bih_dec[col] + bias;
                    for (int j = 0; j < 8; ++j) acc[i][j] = wxv;
                }
            }
            // h @ Whh^T : one A fragment per k, reused for 8 tiles
            for (int k = 0; k < 16; ++k) {
                v16bf a = ld_frag(hB, HH, lane, k * 32);
#pragma unroll
                for (int i = 0; i < 8; ++i) {
                    const int n0 = ((pass * 8 + i) * 8 + wv) * 16;
                    __builtin_prefetch(Whh + (size_t)n0 * HH + (size_t)(lane & 15) * HH
                                       + ((k + 1) & 15) * 32, 0, 1);
                    v16bf b = ld_frag(Whh + (size_t)n0 * HH, HH, lane, k * 32);
                    acc[i] = wmma_bf16(a, b, acc[i]);
                }
            }
            // peephole: c @ Wch^T on i/f/o gate tiles (compile-time predicate)
            for (int k = 0; k < 16; ++k) {
                v16bf a = ld_frag(cB, HH, lane, k * 32);
#pragma unroll
                for (int i = 0; i < 8; ++i) {
                    if (pass == 0 || i >= 4) {
                        const int n0 = ((pass * 8 + i) * 8 + wv) * 16;
                        const int c0 = (n0 < 2 * HH) ? n0 : (n0 - HH);
                        v16bf b = ld_frag(Wch + (size_t)c0 * HH, HH, lane, k * 32);
                        acc[i] = wmma_bf16(a, b, acc[i]);
                    }
                }
            }
            // store to gate buffer
#pragma unroll
            for (int i = 0; i < 8; ++i) {
                const int col = ((pass * 8 + i) * 8 + wv) * 16 + nn;
                for (int j = 0; j < 8; ++j)
                    g4[(mbase + j) * FOURH + col] = acc[i][j];
            }
        }
        __syncthreads();

        // --- gate phase: elementwise update of h, c ---
        for (int idx = tid; idx < 16 * HH; idx += blockDim.x) {
            int r = idx >> 9, hc = idx & (HH - 1);
            float gi = g4[r * FOURH + hc];
            float gf = g4[r * FOURH + HH + hc];
            float gg = g4[r * FOURH + 2 * HH + hc];
            float go = g4[r * FOURH + 3 * HH + hc];
            float i = sigmoid_f(gi);
            float f = sigmoid_f(gf);
            float g = tanhf(gg);
            float o = sigmoid_f(go);
            float cn = f * cS[idx] + i * g;
            float hn = o * tanhf(cn);
            cS[idx] = cn; hS[idx] = hn;
            cB[idx] = f2bf(cn); hB[idx] = f2bf(hn);
            if (hs_out)
                hs_out[((size_t)(b0 + r) * T_steps + t) * HH + hc] = f2bf(hn);
        }
        __syncthreads();
    }

    if (cT_out) {
        for (int idx = tid; idx < 16 * HH; idx += blockDim.x) {
            int r = idx >> 9, hc = idx & (HH - 1);
            cT_out[(b0 + r) * HH + hc] = cS[idx];
        }
    }
}

// ---------------- kernel 4: VAE bottleneck ----------------
__global__ void k_bottleneck(const float* __restrict__ cT,
                             const float* __restrict__ aeW, const float* __restrict__ aeB,
                             const float* __restrict__ muW, const float* __restrict__ muB,
                             const float* __restrict__ stdW, const float* __restrict__ stdB,
                             const float* __restrict__ aedW, const float* __restrict__ aedB,
                             const float* __restrict__ noise,
                             float* __restrict__ mu_t, float* __restrict__ std_t,
                             float* __restrict__ z_out, float* __restrict__ hdec) {
    __shared__ float csS[HH];
    __shared__ float zS[DMU];
    const int row = blockIdx.x, tid = threadIdx.x;

    if (tid < HH) {
        float s = aeB[tid];
        const float* cr = cT + (size_t)row * HH;
        const float* wr = aeW + (size_t)tid * HH;
        for (int k = 0; k < HH; ++k) s += cr[k] * wr[k];
        csS[tid] = softplus_f(s);
    }
    __syncthreads();
    if (tid < DMU) {
        float m = muB[tid], sd = stdB[tid];
        const float* mw = muW + (size_t)tid * HH;
        const float* sw = stdW + (size_t)tid * HH;
        for (int k = 0; k < HH; ++k) { m += csS[k] * mw[k]; sd += csS[k] * sw[k]; }
        float zz = m + expf(0.5f * sd) * noise[row * DMU + tid];
        mu_t[row * DMU + tid] = m;
        std_t[row * DMU + tid] = sd;
        z_out[row * DMU + tid] = zz;
        zS[tid] = zz;
    }
    __syncthreads();
    if (tid < HH) {
        float s = aedB[tid];
        const float* wr = aedW + (size_t)tid * DMU;
        for (int k = 0; k < DMU; ++k) s += zS[k] * wr[k];
        hdec[(size_t)row * HH + tid] = softplus_f(s);
    }
}

// ---------------- kernel 6: reconstruction = hs @ outW^T + out_b ----------------
// M=BT, N=D=128, K=H=512. One wave = 16 rows x all 128 cols (A reused 8x).
__global__ void k_out_gemm(const unsigned short* __restrict__ hsBf,
                           const unsigned short* __restrict__ Wbf,
                           const float* __restrict__ ob,
                           float* __restrict__ recon) {
    const int wv = threadIdx.x >> 5, lane = threadIdx.x & 31;
    const int row0 = (blockIdx.x * 8 + wv) * 16;
    v8f acc[8] = {ZERO8,ZERO8,ZERO8,ZERO8,ZERO8,ZERO8,ZERO8,ZERO8};
    for (int k0 = 0; k0 < HH; k0 += 32) {
        v16bf a = ld_frag(hsBf + (size_t)row0 * HH, HH, lane, k0);
#pragma unroll
        for (int i = 0; i < 8; ++i) {
            v16bf b = ld_frag(Wbf + (size_t)(i * 16) * HH, HH, lane, k0);
            acc[i] = wmma_bf16(a, b, acc[i]);
        }
    }
    const int mbase = (lane >> 4) << 3, nn = lane & 15;
#pragma unroll
    for (int i = 0; i < 8; ++i) {
        const float bv = ob[i * 16 + nn];
        for (int j = 0; j < 8; ++j)
            recon[(size_t)(row0 + mbase + j) * DD + i * 16 + nn] = acc[i][j] + bv;
    }
}

// ---------------- kernel 7: GMM params ----------------
__global__ void k_gmm(const float* __restrict__ mu, const float* __restrict__ var,
                      const float* __restrict__ phi,
                      float* __restrict__ mu_c, float* __restrict__ var_c,
                      float* __restrict__ phi_c) {
    int tid = threadIdx.x;
    for (int i = tid; i < KK * DMU; i += blockDim.x) {
        mu_c[i] = mu[i];
        var_c[i] = softplus_f(var[i]);
    }
    if (tid == 0) {
        float mx = phi[0];
        for (int i = 1; i < KK; ++i) mx = fmaxf(mx, phi[i]);
        float e[KK], s = 0.f;
        for (int i = 0; i < KK; ++i) { e[i] = expf(phi[i] - mx); s += e[i]; }
        for (int i = 0; i < KK; ++i) phi_c[i] = e[i] / s;
    }
}

// ---------------- host launcher ----------------
extern "C" void kernel_launch(void* const* d_in, const int* in_sizes, int n_in,
                              void* d_out, int out_size, void* d_ws, size_t ws_size,
                              hipStream_t stream) {
    const float* X        = (const float*)d_in[0];
    const float* mask     = (const float*)d_in[1];
    const float* noise    = (const float*)d_in[2];
    const float* imp_W    = (const float*)d_in[3];
    const float* enc_Wih  = (const float*)d_in[4];
    const float* enc_Whh  = (const float*)d_in[5];
    const float* enc_bih  = (const float*)d_in[6];
    const float* enc_bhh  = (const float*)d_in[7];
    const float* enc_Wch  = (const float*)d_in[8];
    const float* enc_bch  = (const float*)d_in[9];
    // d_in[10] = dec_Wih (unused: decoder inputs are zero)
    const float* dec_Whh  = (const float*)d_in[11];
    const float* dec_bih  = (const float*)d_in[12];
    const float* dec_bhh  = (const float*)d_in[13];
    const float* dec_Wch  = (const float*)d_in[14];
    const float* dec_bch  = (const float*)d_in[15];
    const float* ae_enc_W = (const float*)d_in[16];
    const float* ae_enc_b = (const float*)d_in[17];
    const float* ae_dec_W = (const float*)d_in[18];
    const float* ae_dec_b = (const float*)d_in[19];
    const float* mu_W     = (const float*)d_in[20];
    const float* mu_b     = (const float*)d_in[21];
    const float* std_W    = (const float*)d_in[22];
    const float* std_b    = (const float*)d_in[23];
    const float* out_W    = (const float*)d_in[24];
    const float* out_b    = (const float*)d_in[25];
    const float* gmm_mu   = (const float*)d_in[26];
    const float* gmm_var  = (const float*)d_in[27];
    const float* gmm_phi  = (const float*)d_in[28];

    // ---- output layout (fp32, concatenated in return order) ----
    float* out     = (float*)d_out;
    float* recon   = out;                            // B*T*D = 4194304
    float* mu_c    = recon + (size_t)BT * DD;        // K*DMU = 512
    float* var_c   = mu_c + KK * DMU;                // 512
    float* phi_c   = var_c + KK * DMU;               // 8
    float* z_out   = phi_c + KK;                     // B*DMU = 16384
    float* mu_t    = z_out + BB * DMU;               // 16384
    float* std_t   = mu_t + BB * DMU;                // 16384

    // ---- workspace layout ----
    char* base = (char*)d_ws;
    size_t o = 0;
    auto alloc = [&](size_t bytes) { size_t r = o; o += (bytes + 255) & ~(size_t)255; return r; };
    unsigned short* Xbf      = (unsigned short*)(base + alloc((size_t)BT * DD * 2));
    unsigned short* impWbf   = (unsigned short*)(base + alloc((size_t)DD * DD * 2));
    unsigned short* encWihbf = (unsigned short*)(base + alloc((size_t)FOURH * DD * 2));
    unsigned short* encWhhbf = (unsigned short*)(base + alloc((size_t)FOURH * HH * 2));
    unsigned short* encWchbf = (unsigned short*)(base + alloc((size_t)THREEH * HH * 2));
    unsigned short* decWhhbf = (unsigned short*)(base + alloc((size_t)FOURH * HH * 2));
    unsigned short* decWchbf = (unsigned short*)(base + alloc((size_t)THREEH * HH * 2));
    unsigned short* outWbf   = (unsigned short*)(base + alloc((size_t)DD * HH * 2));
    unsigned short* XimpBf   = (unsigned short*)(base + alloc((size_t)BT * DD * 2));
    float*          wx_all   = (float*)(base + alloc((size_t)BT * FOURH * 4));
    unsigned short* hsBf     = (unsigned short*)(base + alloc((size_t)BT * HH * 2));
    float*          cT       = (float*)(base + alloc((size_t)BB * HH * 4));
    float*          hdec     = (float*)(base + alloc((size_t)BB * HH * 4));

    auto conv = [&](const float* s, unsigned short* d, int n) {
        k_f32_to_bf16<<<(n + 255) / 256, 256, 0, stream>>>(s, d, n);
    };
    conv(X, Xbf, BT * DD);
    conv(imp_W, impWbf, DD * DD);
    conv(enc_Wih, encWihbf, FOURH * DD);
    conv(enc_Whh, encWhhbf, FOURH * HH);
    conv(enc_Wch, encWchbf, THREEH * HH);
    conv(dec_Whh, decWhhbf, FOURH * HH);
    conv(dec_Wch, decWchbf, THREEH * HH);
    conv(out_W, outWbf, DD * HH);

    // imputation: 2048 m-tiles / 8 waves = 256 blocks (each wave: 16x128)
    k_imp_gemm<<<256, 256, 0, stream>>>(Xbf, impWbf, X, mask, XimpBf);

    // wx_all: 2048 m-tiles x 16 n-groups = 32768 jobs / 8 = 4096 blocks
    k_wx_gemm<<<4096, 256, 0, stream>>>(XimpBf, encWihbf, enc_bih, wx_all);

    const size_t smem = (size_t)16 * HH * 4 * 2   // hS, cS
                      + (size_t)16 * HH * 2 * 2   // hB, cB
                      + (size_t)16 * FOURH * 4;   // g4  => 229376 bytes
    // encoder recurrence: 16 blocks (16 batch rows each), all T steps inside
    k_lstm_rec<<<16, 256, smem, stream>>>(encWhhbf, encWchbf, enc_bhh, enc_bch,
                                          wx_all, nullptr, nullptr,
                                          cT, nullptr, TT);

    // VAE bottleneck
    k_bottleneck<<<BB, HH, 0, stream>>>(cT, ae_enc_W, ae_enc_b,
                                        mu_W, mu_b, std_W, std_b,
                                        ae_dec_W, ae_dec_b, noise,
                                        mu_t, std_t, z_out, hdec);

    // decoder recurrence (wx = dec_bih broadcast)
    k_lstm_rec<<<16, 256, smem, stream>>>(decWhhbf, decWchbf, dec_bhh, dec_bch,
                                          nullptr, dec_bih, hdec,
                                          nullptr, hsBf, TT);

    // reconstruction: 2048 m-tiles / 8 waves = 256 blocks
    k_out_gemm<<<256, 256, 0, stream>>>(hsBf, outWbf, out_b, recon);

    // GMM params
    k_gmm<<<1, 256, 0, stream>>>(gmm_mu, gmm_var, gmm_phi, mu_c, var_c, phi_c);
}
